// LocalTriangleAttentionNew_71279277244716
// MI455X (gfx1250) — matrix operand order repaired and checked by reading, not compiled, not run on hardware
//
#include <hip/hip_runtime.h>

#define NTOK 384
#define CZ   128
#define ME   (NTOK * NTOK)   // 147456 pair rows
#define KNBR 32
#define MG   (NTOK * KNBR)   // 12288 gathered rows
#define HINF 1e9f
#define WMAT 16384           // 128*128 weight matrix elements

typedef __attribute__((ext_vector_type(16))) __bf16 v16bf;
typedef __attribute__((ext_vector_type(8)))  float  v8f;

__device__ __forceinline__ __bf16 f2bf(float f) {
  unsigned u = __builtin_bit_cast(unsigned, f);
  u += 0x7FFFu + ((u >> 16) & 1u);           // round-to-nearest-even
  unsigned short h = (unsigned short)(u >> 16);
  return __builtin_bit_cast(__bf16, h);
}
__device__ __forceinline__ float bf2f(__bf16 b) {
  unsigned short h = __builtin_bit_cast(unsigned short, b);
  unsigned u = ((unsigned)h) << 16;
  return __builtin_bit_cast(float, u);
}
__device__ __forceinline__ float sigf(float x) { return 1.f / (1.f + __expf(-x)); }

// one-time f32 -> bf16 weight conversion (128x128 matrix)
__global__ __launch_bounds__(256)
void convert_w(const float* __restrict__ src, __bf16* __restrict__ dst) {
  const int i = blockIdx.x * 256 + threadIdx.x;   // grid = 16 blocks
#pragma unroll
  for (int r = 0; r < 4; ++r) dst[i + r * 4096] = f2bf(src[i + r * 4096]);
}

// ---------------------------------------------------------------------------
// Generic K=128 -> Nout=128 GEMM with WMMA bf16. One block = 16 output rows,
// 8 waves cover all 128 output columns (one 16x16 tile per wave).
// A: bf16 [M,128] row-major. W: bf16 [128,128] (out-channel major, pre-converted).
// MODE is a template parameter so each epilogue variant is branch-free and the
// dual-accumulator (DUALMUL) path issues back-to-back WMMAs.
// ---------------------------------------------------------------------------
enum {
  GM_BF16 = 0,         // outB = P (*scale, +B1)
  GM_F32,              // outF = P (*scale)
  GM_DUALMUL,          // outB = rowmask * (P+B1) * sigmoid(P2+B2)
  GM_SIGMOID_BF16,     // outB = sigmoid(P+B1)
  GM_SIGMOID_F32,      // outF = sigmoid(P+B1)
  GM_GATED_ADD,        // outF[m,n] += (P+B1) * bf2f(gmul[m,n])
  GM_SCATTER_ADD       // outF[tgt(m),n] += (P+B1), tgt via idx (kNN scatter)
};

template <int MODE>
__global__ __launch_bounds__(256)
void gemm128(const __bf16* __restrict__ A, const __bf16* __restrict__ W1,
             const float* __restrict__ B1, const __bf16* __restrict__ W2,
             const float* __restrict__ B2, const float* __restrict__ rowmask,
             const __bf16* __restrict__ gmul, const int* __restrict__ idx,
             float* __restrict__ outF, __bf16* __restrict__ outB,
             float scale, int scatter_end)
{
  __shared__ __bf16 lds_a[16 * 128];        // 4 KB A tile (16 rows x full K)
  const int m0 = blockIdx.x * 16;
  const int t = threadIdx.x;
  { // cooperative stage: 4096 B, 16 B per thread
    const uint4* src = (const uint4*)(A + (size_t)m0 * CZ);
    ((uint4*)lds_a)[t] = src[t];
  }
  __syncthreads();

  const int w = t >> 5;                 // wave id -> n-tile
  const int lane = t & 31;
  const int lhalf = lane >> 4;          // 0 | 1
  const int lmod = lane & 15;
  const int n0 = w * 16;

  v8f acc  = {0.f,0.f,0.f,0.f,0.f,0.f,0.f,0.f};
  v8f acc2 = {0.f,0.f,0.f,0.f,0.f,0.f,0.f,0.f};

#pragma unroll
  for (int kc = 0; kc < 4; ++kc) {
    // A fragment: lane holds row lmod; K = kc*32 + lhalf*16 + {0..15}
    union { uint4 u[2]; v16bf v; } af;
    const uint4* ap = (const uint4*)(lds_a + lmod * 128 + kc * 32 + lhalf * 16);
    af.u[0] = ap[0];
    af.u[1] = ap[1];
    // W fragment: lane holds col n0+lmod; K = kc*32 + lhalf*16 + e (contiguous,
    // pre-converted bf16 -> two 16B vector loads)
    {
      const uint4* wp = (const uint4*)(W1 + (size_t)(n0 + lmod) * 128 +
                                       kc * 32 + lhalf * 16);
      union { uint4 u[2]; v16bf v; } wf;
      wf.u[0] = wp[0]; wf.u[1] = wp[1];
      acc = __builtin_amdgcn_wmma_f32_16x16x32_bf16(false, af.v, false, wf.v,
                                                    (short)0, acc, false, false);
    }
    if constexpr (MODE == GM_DUALMUL) {
      const uint4* wp2 = (const uint4*)(W2 + (size_t)(n0 + lmod) * 128 +
                                        kc * 32 + lhalf * 16);
      union { uint4 u[2]; v16bf v; } wf2;
      wf2.u[0] = wp2[0]; wf2.u[1] = wp2[1];
      acc2 = __builtin_amdgcn_wmma_f32_16x16x32_bf16(false, af.v, false, wf2.v,
                                                     (short)0, acc2, false, false);
    }
  }

  const int n = n0 + lmod;
#pragma unroll
  for (int r = 0; r < 8; ++r) {
    const int m = m0 + r + lhalf * 8;   // C/D layout: lanes16-31 hold M+8
    float v = acc[r] * scale;
    if (B1) v += B1[n];
    if constexpr (MODE == GM_BF16) {
      outB[(size_t)m * CZ + n] = f2bf(v);
    } else if constexpr (MODE == GM_F32) {
      outF[(size_t)m * CZ + n] = v;
    } else if constexpr (MODE == GM_DUALMUL) {
      float g = acc2[r] + (B2 ? B2[n] : 0.f);
      float mk = rowmask ? rowmask[m] : 1.f;
      outB[(size_t)m * CZ + n] = f2bf(mk * v * sigf(g));
    } else if constexpr (MODE == GM_SIGMOID_BF16) {
      outB[(size_t)m * CZ + n] = f2bf(sigf(v));
    } else if constexpr (MODE == GM_SIGMOID_F32) {
      outF[(size_t)m * CZ + n] = sigf(v);
    } else if constexpr (MODE == GM_GATED_ADD) {
      outF[(size_t)m * CZ + n] += v * bf2f(gmul[(size_t)m * CZ + n]);
    } else if constexpr (MODE == GM_SCATTER_ADD) {
      int i = m >> 5;
      int j = idx[m];
      size_t tgt = scatter_end ? ((size_t)j * NTOK + i) : ((size_t)i * NTOK + j);
      outF[tgt * CZ + n] += v;
    }
  }
}

// ---------------------------------------------------------------------------
// Triangle einsum: X[i,j,c] = sum_k a[i,k,c]*b[j,k,c]   (outgoing)
//                  X[i,j,c] = sum_k a[k,i,c]*b[k,j,c]   (incoming)
// 128 per-channel 384x384x384 matmuls. Block: one (i,j) 16x16 tile x 8 channels
// (one channel per wave). LDS tiles stored [c][row][k] for vector frag reads.
// ---------------------------------------------------------------------------
__global__ __launch_bounds__(256)
void tri_einsum(const __bf16* __restrict__ a, const __bf16* __restrict__ b,
                float* __restrict__ X, int outgoing)
{
  __shared__ __bf16 a_lds[8][16][32];   // 8 KB
  __shared__ __bf16 b_lds[8][16][32];   // 8 KB
  const int i0 = blockIdx.x * 16, j0 = blockIdx.y * 16, c0 = blockIdx.z * 8;
  const int t = threadIdx.x;
  const int w = t >> 5, lane = t & 31;
  const int lhalf = lane >> 4, lmod = lane & 15;

  v8f acc = {0.f,0.f,0.f,0.f,0.f,0.f,0.f,0.f};

  for (int kc = 0; kc < 12; ++kc) {
    const int k0 = kc * 32;
    __syncthreads();
#pragma unroll
    for (int rep = 0; rep < 2; ++rep) {
      const int lin = t + rep * 256;            // 512 (row,k) positions
      const int rr = lin >> 5, kk = lin & 31;
      size_t ga = outgoing ? ((size_t)(i0 + rr) * NTOK + (k0 + kk))
                           : ((size_t)(k0 + kk) * NTOK + (i0 + rr));
      union { uint4 u; __bf16 h[8]; } va;
      va.u = *(const uint4*)(a + ga * CZ + c0); // 8 channels, 16 B
#pragma unroll
      for (int ch = 0; ch < 8; ++ch) a_lds[ch][rr][kk] = va.h[ch];
      size_t gb = outgoing ? ((size_t)(j0 + rr) * NTOK + (k0 + kk))
                           : ((size_t)(k0 + kk) * NTOK + (j0 + rr));
      union { uint4 u; __bf16 h[8]; } vb;
      vb.u = *(const uint4*)(b + gb * CZ + c0);
#pragma unroll
      for (int ch = 0; ch < 8; ++ch) b_lds[ch][rr][kk] = vb.h[ch];
      if (kc < 11) {                            // prefetch next K chunk
        size_t pa = outgoing ? ((size_t)(i0 + rr) * NTOK + (k0 + 32 + kk))
                             : ((size_t)(k0 + 32 + kk) * NTOK + (i0 + rr));
        __builtin_prefetch(a + pa * CZ + c0, 0, 3);
        size_t pb = outgoing ? ((size_t)(j0 + rr) * NTOK + (k0 + 32 + kk))
                             : ((size_t)(k0 + 32 + kk) * NTOK + (j0 + rr));
        __builtin_prefetch(b + pb * CZ + c0, 0, 3);
      }
    }
    __syncthreads();

    union { uint4 u[2]; v16bf v; } af, bf_;
    const uint4* ap = (const uint4*)(&a_lds[w][lmod][lhalf * 16]);
    af.u[0] = ap[0]; af.u[1] = ap[1];                 // 16 consecutive K
    const uint4* bp = (const uint4*)(&b_lds[w][lmod][lhalf * 16]);
    bf_.u[0] = bp[0]; bf_.u[1] = bp[1];               // 16 consecutive K
    acc = __builtin_amdgcn_wmma_f32_16x16x32_bf16(false, af.v, false, bf_.v,
                                                  (short)0, acc, false, false);
  }

  const int j = j0 + lmod;
#pragma unroll
  for (int r = 0; r < 8; ++r) {
    const int i = i0 + r + lhalf * 8;
    X[((size_t)i * NTOK + j) * CZ + c0 + w] = acc[r];
  }
}

// --------------------------- LayerNorm (C=128) -----------------------------
__global__ __launch_bounds__(128)
void ln128(const float* __restrict__ x, const float* __restrict__ s,
           const float* __restrict__ b, __bf16* __restrict__ out)
{
  __shared__ float red[128];
  const int row = blockIdx.x, c = threadIdx.x;
  float v = x[(size_t)row * CZ + c];
  red[c] = v; __syncthreads();
  for (int o = 64; o > 0; o >>= 1) { if (c < o) red[c] += red[c + o]; __syncthreads(); }
  float mu = red[0] * (1.f / 128.f); __syncthreads();
  float d = v - mu;
  red[c] = d * d; __syncthreads();
  for (int o = 64; o > 0; o >>= 1) { if (c < o) red[c] += red[c + o]; __syncthreads(); }
  float inv = rsqrtf(red[0] * (1.f / 128.f) + 1e-5f);
  out[(size_t)row * CZ + c] = f2bf(d * inv * s[c] + b[c]);
}

// gather kNN neighbor rows of z and LayerNorm them -> bf16 [MG,128]
__global__ __launch_bounds__(128)
void gather_ln(const float* __restrict__ z, const int* __restrict__ idx,
               const float* __restrict__ s, const float* __restrict__ b,
               __bf16* __restrict__ out, int ending)
{
  __shared__ float red[128];
  const int r = blockIdx.x, c = threadIdx.x;
  const int i = r >> 5, j = idx[r];
  const size_t src = ending ? ((size_t)j * NTOK + i) : ((size_t)i * NTOK + j);
  float v = z[src * CZ + c];
  red[c] = v; __syncthreads();
  for (int o = 64; o > 0; o >>= 1) { if (c < o) red[c] += red[c + o]; __syncthreads(); }
  float mu = red[0] * (1.f / 128.f); __syncthreads();
  float d = v - mu;
  red[c] = d * d; __syncthreads();
  for (int o = 64; o > 0; o >>= 1) { if (c < o) red[c] += red[c + o]; __syncthreads(); }
  float inv = rsqrtf(red[0] * (1.f / 128.f) + 1e-5f);
  out[(size_t)r * CZ + c] = f2bf(d * inv * s[c] + b[c]);
}

// --------------------- node 16-dim left/right projections ------------------
__global__ __launch_bounds__(32)
void node_proj(const float* __restrict__ ne,
               const float* __restrict__ wl, const float* __restrict__ bl,
               const float* __restrict__ wr, const float* __restrict__ br,
               float* __restrict__ left, float* __restrict__ right)
{
  const int i = blockIdx.x, t = threadIdx.x;
  const float* x = ne + (size_t)i * 384;
  if (t < 16) {
    float acc = bl[t];
    for (int s = 0; s < 384; ++s) acc += x[s] * wl[t * 384 + s];
    left[i * 16 + t] = acc;
  } else {
    const int u = t - 16;
    float acc = br[u];
    for (int s = 0; s < 384; ++s) acc += x[s] * wr[u * 384 + s];
    right[i * 16 + u] = acc;
  }
}

// ---------------- per-pair RBF + outer-product gate -> 4-head bias ---------
__global__ __launch_bounds__(128)
void bias_kernel(const float* __restrict__ coords, const float* __restrict__ left,
                 const float* __restrict__ right, const float* __restrict__ rbf_w,
                 const float* __restrict__ rbf_b, const float* __restrict__ gate_w,
                 const float* __restrict__ gate_b, const float* __restrict__ tb_w,
                 float* __restrict__ bias)
{
  __shared__ float lf[16], rt[16], red[128];
  const int pidx = blockIdx.x, i = pidx / NTOK, j = pidx % NTOK;
  const int c = threadIdx.x;
  if (c < 16) lf[c] = left[i * 16 + c];
  else if (c < 32) rt[c - 16] = right[j * 16 + (c - 16)];
  __syncthreads();
  float dx = coords[i * 3] - coords[j * 3];
  float dy = coords[i * 3 + 1] - coords[j * 3 + 1];
  float dz = coords[i * 3 + 2] - coords[j * 3 + 2];
  float d = sqrtf(dx * dx + dy * dy + dz * dz + 1e-12f);
  float y1 = rbf_b[c];
  for (int r2 = 0; r2 < 64; ++r2) {
    float t2 = (d - 0.5f * r2) * 2.0f;       // (d-mu)/0.5
    y1 += rbf_w[c * 64 + r2] * __expf(-t2 * t2);
  }
  float gi = gate_b[c];
  for (int pq = 0; pq < 256; ++pq)
    gi += gate_w[c * 256 + pq] * lf[pq >> 4] * rt[pq & 15];
  float v = y1 * sigf(gi);
  for (int h = 0; h < 4; ++h) {
    red[c] = v * tb_w[h * 128 + c];
    __syncthreads();
    for (int o = 64; o > 0; o >>= 1) { if (c < o) red[c] += red[c + o]; __syncthreads(); }
    if (c == 0) bias[(size_t)pidx * 4 + h] = red[0];
    __syncthreads();
  }
}

// --------------------------- kNN index selection ---------------------------
__global__ __launch_bounds__(128)
void knn_kernel(const float* __restrict__ coords, const float* __restrict__ mask,
                int* __restrict__ idx, int ending)
{
  __shared__ float dist[NTOK];
  const int i = blockIdx.x, t = threadIdx.x;
  const float xi = coords[i * 3], yi = coords[i * 3 + 1], zi = coords[i * 3 + 2];
  for (int j = t; j < NTOK; j += 128) {
    float dx = xi - coords[j * 3], dy = yi - coords[j * 3 + 1], dz = zi - coords[j * 3 + 2];
    float d = sqrtf(dx * dx + dy * dy + dz * dz + 1e-12f);
    if (j == i) d = HINF;                        // exclude self
    else { int ad = j - i; if (ad < 0) ad = -ad; if (ad <= 4) d = 0.f; }  // band
    float m = ending ? mask[(size_t)j * NTOK + i] : mask[(size_t)i * NTOK + j];
    dist[j] = d + HINF * (m - 1.f);
  }
  __syncthreads();
  if (t == 0) {                                  // stable 32-way selection
    for (int s = 0; s < KNBR; ++s) {
      float best = dist[0]; int bj = 0;
      for (int j = 1; j < NTOK; ++j) if (dist[j] < best) { best = dist[j]; bj = j; }
      idx[i * KNBR + s] = bj;
      dist[bj] = 3e9f;
    }
  }
}

// ------------- local attention core: one (row i, head h) per wave ----------
__global__ __launch_bounds__(32)
void attn_kernel(const float* __restrict__ q, const float* __restrict__ k,
                 const float* __restrict__ v, const float* __restrict__ g,
                 const float* __restrict__ bias, const float* __restrict__ mask,
                 const int* __restrict__ idx, __bf16* __restrict__ outb, int ending)
{
  __shared__ float kv[32][32], vv[32][32], bb[32];
  const int i = blockIdx.x, h = blockIdx.y;
  const int kk = threadIdx.x;
  {
    const size_t rk = (size_t)(i * KNBR + kk) * CZ + h * 32;
    for (int c = 0; c < 32; ++c) { kv[kk][c] = k[rk + c]; vv[kk][c] = v[rk + c]; }
    const int j = idx[i * KNBR + kk];
    float m = ending ? mask[(size_t)j * NTOK + i] : mask[(size_t)i * NTOK + j];
    float tb = ending ? bias[((size_t)j * NTOK + i) * 4 + h]
                      : bias[((size_t)i * NTOK + j) * 4 + h];
    bb[kk] = HINF * (m - 1.f) + tb;
  }
  __syncthreads();
  const int qq = threadIdx.x;
  const size_t rq = (size_t)(i * KNBR + qq) * CZ + h * 32;
  float qr[32];
  for (int c = 0; c < 32; ++c) qr[c] = q[rq + c];   // already scaled by 1/sqrt(32)
  float logits[32]; float mx = -1e30f;
  for (int t2 = 0; t2 < 32; ++t2) {
    float s = 0.f;
    for (int c = 0; c < 32; ++c) s += qr[c] * kv[t2][c];
    s += bb[t2]; logits[t2] = s; mx = fmaxf(mx, s);
  }
  float den = 0.f;
  for (int t2 = 0; t2 < 32; ++t2) { logits[t2] = __expf(logits[t2] - mx); den += logits[t2]; }
  const float inv = 1.f / den;
  for (int c = 0; c < 32; ++c) {
    float o = 0.f;
    for (int t2 = 0; t2 < 32; ++t2) o += logits[t2] * vv[t2][c];
    outb[rq + c] = f2bf(o * inv * g[rq + c]);
  }
}

// ===========================================================================
extern "C" void kernel_launch(void* const* d_in, const int* in_sizes, int n_in,
                              void* d_out, int out_size, void* d_ws, size_t ws_size,
                              hipStream_t stream) {
  (void)in_sizes; (void)n_in; (void)out_size; (void)ws_size;
  const float* node   = (const float*)d_in[0];
  const float* edge   = (const float*)d_in[1];
  const float* coords = (const float*)d_in[2];
  const float* mask   = (const float*)d_in[3];
  const float* p[57];
  for (int i = 0; i < 57; ++i) p[i] = (const float*)d_in[4 + i];
  // params flat order: 0..10 top-level, 11..26 tmo, 27..42 tmi, 43..49 mha_start, 50..56 mha_end
  enum { P_PLW=0,P_PLB,P_PRW,P_PRB,P_TGW,P_TGB,P_RBW,P_RBB,P_TBW,P_LNS,P_LNB };
  enum { T_LNIS=0,T_LNIB,T_APW,T_APB,T_AGW,T_AGB,T_BPW,T_BPB,T_BGW,T_BGB,
         T_GW,T_GB,T_LNOS,T_LNOB,T_ZW,T_ZB };
  enum { A_QW=0,A_KW,A_VW,A_GW,A_GB,A_OW,A_OB };

  // workspace carve (~191 MB)
  size_t off = 0;
  auto carve = [&](size_t bytes) {
    char* q = (char*)d_ws + off; off += (bytes + 255) & ~(size_t)255; return (void*)q;
  };
  __bf16* bufZ = (__bf16*)carve((size_t)ME * CZ * 2);   // LN(z) / gathered LN
  __bf16* bufA = (__bf16*)carve((size_t)ME * CZ * 2);   // a / tri-gate / attn out
  __bf16* bufB = (__bf16*)carve((size_t)ME * CZ * 2);   // b / LN(x)
  float*  bufX = (float*)carve((size_t)ME * CZ * 4);    // einsum out / q,k,v,g
  float*  biasb = (float*)carve((size_t)ME * 4 * 4);
  float*  leftb = (float*)carve(NTOK * 16 * 4);
  float*  rightb = (float*)carve(NTOK * 16 * 4);
  int*    idxb = (int*)carve((size_t)MG * 4);
  __bf16* wcvt = (__bf16*)carve((size_t)22 * WMAT * 2); // pre-converted weights
  float* qb = bufX;
  float* kb = bufX + (size_t)MG * CZ;
  float* vb = bufX + 2 * (size_t)MG * CZ;
  float* gmb = bufX + 3 * (size_t)MG * CZ;
  float* z = (float*)d_out;

  // ---- one-time bf16 conversion of all 22 WMMA weight matrices ----
  // slots: tri pass0 {ap,ag,bp,bg,g,z}=0..5, tri pass1 = 6..11,
  //        mha pass0 {q,k,v,g,o}=12..16, mha pass1 = 17..21
  {
    const float* wsrc[22]; int si = 0;
    for (int pass = 0; pass < 2; ++pass) {
      const float* const* tp = p + (pass == 0 ? 11 : 27);
      wsrc[si++] = tp[T_APW]; wsrc[si++] = tp[T_AGW]; wsrc[si++] = tp[T_BPW];
      wsrc[si++] = tp[T_BGW]; wsrc[si++] = tp[T_GW];  wsrc[si++] = tp[T_ZW];
    }
    for (int pass = 0; pass < 2; ++pass) {
      const float* const* ap2 = p + (pass == 0 ? 43 : 50);
      wsrc[si++] = ap2[A_QW]; wsrc[si++] = ap2[A_KW]; wsrc[si++] = ap2[A_VW];
      wsrc[si++] = ap2[A_GW]; wsrc[si++] = ap2[A_OW];
    }
    for (int s2 = 0; s2 < 22; ++s2)
      convert_w<<<16, 256, 0, stream>>>(wsrc[s2], wcvt + (size_t)s2 * WMAT);
  }

  // z = edge_embed
  hipMemcpyAsync(z, edge, (size_t)ME * CZ * 4, hipMemcpyDeviceToDevice, stream);
  // bias path
  node_proj<<<NTOK, 32, 0, stream>>>(node, p[P_PLW], p[P_PLB], p[P_PRW], p[P_PRB],
                                     leftb, rightb);
  bias_kernel<<<ME, 128, 0, stream>>>(coords, leftb, rightb, p[P_RBW], p[P_RBB],
                                      p[P_TGW], p[P_TGB], p[P_TBW], biasb);

  // ---- two triangle multiplications (outgoing, incoming) ----
  for (int pass = 0; pass < 2; ++pass) {
    const float* const* tp = p + (pass == 0 ? 11 : 27);
    const __bf16* wb = wcvt + (size_t)pass * 6 * WMAT;
    const int outgoing = (pass == 0);
    ln128<<<ME, 128, 0, stream>>>(z, tp[T_LNIS], tp[T_LNIB], bufZ);
    gemm128<GM_DUALMUL><<<ME / 16, 256, 0, stream>>>(bufZ, wb + 0 * WMAT,
        tp[T_APB], wb + 1 * WMAT, tp[T_AGB], mask, nullptr, nullptr,
        nullptr, bufA, 1.f, 0);
    gemm128<GM_DUALMUL><<<ME / 16, 256, 0, stream>>>(bufZ, wb + 2 * WMAT,
        tp[T_BPB], wb + 3 * WMAT, tp[T_BGB], mask, nullptr, nullptr,
        nullptr, bufB, 1.f, 0);
    dim3 gr(NTOK / 16, NTOK / 16, CZ / 8);
    tri_einsum<<<gr, 256, 0, stream>>>(bufA, bufB, bufX, outgoing);
    gemm128<GM_SIGMOID_BF16><<<ME / 16, 256, 0, stream>>>(bufZ, wb + 4 * WMAT,
        tp[T_GB], nullptr, nullptr, nullptr, nullptr, nullptr,
        nullptr, bufA, 1.f, 0);
    ln128<<<ME, 128, 0, stream>>>(bufX, tp[T_LNOS], tp[T_LNOB], bufB);
    gemm128<GM_GATED_ADD><<<ME / 16, 256, 0, stream>>>(bufB, wb + 5 * WMAT,
        tp[T_ZB], nullptr, nullptr, nullptr, bufA, nullptr,
        z, nullptr, 1.f, 0);
  }

  // ---- two local kNN attentions (starting, ending) ----
  for (int pass = 0; pass < 2; ++pass) {
    const float* const* ap = p + (pass == 0 ? 43 : 50);
    const __bf16* wb = wcvt + (size_t)(12 + pass * 5) * WMAT;
    const int ending = pass;
    knn_kernel<<<NTOK, 128, 0, stream>>>(coords, mask, idxb, ending);
    gather_ln<<<MG, 128, 0, stream>>>(z, idxb, p[P_LNS], p[P_LNB], bufZ, ending);
    gemm128<GM_F32><<<MG / 16, 256, 0, stream>>>(bufZ, wb + 0 * WMAT, nullptr,
        nullptr, nullptr, nullptr, nullptr, nullptr, qb, nullptr,
        0.17677669529663687f, 0);
    gemm128<GM_F32><<<MG / 16, 256, 0, stream>>>(bufZ, wb + 1 * WMAT, nullptr,
        nullptr, nullptr, nullptr, nullptr, nullptr, kb, nullptr, 1.f, 0);
    gemm128<GM_F32><<<MG / 16, 256, 0, stream>>>(bufZ, wb + 2 * WMAT, nullptr,
        nullptr, nullptr, nullptr, nullptr, nullptr, vb, nullptr, 1.f, 0);
    gemm128<GM_SIGMOID_F32><<<MG / 16, 256, 0, stream>>>(bufZ, wb + 3 * WMAT,
        ap[A_GB], nullptr, nullptr, nullptr, nullptr, nullptr,
        gmb, nullptr, 1.f, 0);
    attn_kernel<<<dim3(NTOK, 4), 32, 0, stream>>>(qb, kb, vb, gmb, biasb, mask,
                                                  idxb, bufA, ending);
    gemm128<GM_SCATTER_ADD><<<MG / 16, 256, 0, stream>>>(bufA, wb + 4 * WMAT,
        ap[A_OB], nullptr, nullptr, nullptr, nullptr, idxb,
        z, nullptr, 1.f, ending);
  }
}
